// SelfAttentionHead_13889924235942
// MI455X (gfx1250) — matrix operand order repaired and checked by reading, not compiled, not run on hardware
//
#include <hip/hip_runtime.h>
#include <hip/hip_bf16.h>

typedef __bf16 bf16;
typedef __attribute__((ext_vector_type(16))) __bf16 v16bf;
typedef __attribute__((ext_vector_type(8)))  __bf16 v8bf;
typedef __attribute__((ext_vector_type(8)))  float  v8f;

constexpr int   Bn = 4, Tn = 4096, En = 1024, Dn = 64;
constexpr float SCALE  = 0.125f;                 // 1/sqrt(64)
constexpr float LOG2E  = 1.44269504088896340736f;
constexpr float NEGINF = -1e30f;

__device__ __forceinline__ v8f wmma_bf16(v16bf a, v16bf b, v8f c) {
  return __builtin_amdgcn_wmma_f32_16x16x32_bf16(false, a, false, b, (short)0, c,
                                                 false, false);
}

// ---------------------------------------------------------------------------
// Kernel 1: convert + transpose weights to bf16, Wt[mat][n][k] layout so that
// WMMA B-fragments (K contiguous per lane) are single 32B loads.
// ---------------------------------------------------------------------------
__global__ __launch_bounds__(256) void wcvt_kernel(const float* __restrict__ Wk,
                                                   const float* __restrict__ Wq,
                                                   const float* __restrict__ Wv,
                                                   bf16* __restrict__ Wt) {
  int idx = blockIdx.x * 256 + threadIdx.x;      // 3 * 64 * 1024 total
  if (idx >= 3 * Dn * En) return;
  int mat = idx / (Dn * En);
  int rem = idx - mat * (Dn * En);
  int n   = rem / En;
  int k   = rem - n * En;
  const float* W = (mat == 0) ? Wk : (mat == 1 ? Wq : Wv);
  Wt[idx] = (bf16)W[k * Dn + n];
}

// ---------------------------------------------------------------------------
// Kernel 2: Q/K/V projections.  One wave = one 16-row M tile of one matrix.
// D = A(16x32 bf16) x B(32x16 bf16) + C, accumulated over E/32 = 32 k-steps.
// All 4 B fragments preloaded per k-step so the 4 WMMAs issue back-to-back.
// ---------------------------------------------------------------------------
__global__ __launch_bounds__(128) void proj_kernel(const float* __restrict__ emb,
                                                   const float* __restrict__ bk,
                                                   const float* __restrict__ bq,
                                                   const float* __restrict__ bv,
                                                   const bf16*  __restrict__ Wt,
                                                   bf16*        __restrict__ qkv) {
  const int lane = threadIdx.x & 31;
  const int wave = threadIdx.x >> 5;
  const int h    = lane >> 4;        // lane half: selects K sub-range
  const int n    = lane & 15;        // A: row M;  B/C: column N
  const int mat  = blockIdx.y;       // 0=K, 1=Q, 2=V (input order)
  const int row0 = (blockIdx.x * 4 + wave) * 16;

  const float* bias = (mat == 0) ? bk : (mat == 1 ? bq : bv);
  const bf16*  W    = Wt  + (size_t)mat * Dn * En;
  bf16*        outp = qkv + (size_t)mat * (size_t)(Bn * Tn) * Dn;

  v8f acc[4];
#pragma unroll
  for (int t = 0; t < 4; ++t)
#pragma unroll
    for (int r = 0; r < 8; ++r) acc[t][r] = 0.f;

  const float* arow = emb + (size_t)(row0 + n) * En;
  for (int kk = 0; kk < 32; ++kk) {
    const int k0 = kk * 32;
    // Preload all B fragments for this k-step (clause-friendly).
    v16bf bfr[4];
#pragma unroll
    for (int t = 0; t < 4; ++t)
      bfr[t] = *(const v16bf*)(W + (size_t)(t * 16 + n) * En + k0 + 16 * h);
    // A fragment: one contiguous run of 16 fp32, convert to bf16.
    const float4 f0 = *(const float4*)(arow + k0 + 16 * h);
    const float4 f1 = *(const float4*)(arow + k0 + 16 * h + 4);
    const float4 f2 = *(const float4*)(arow + k0 + 16 * h + 8);
    const float4 f3 = *(const float4*)(arow + k0 + 16 * h + 12);
    v16bf a;
    a[0] = (bf16)f0.x;  a[1] = (bf16)f0.y;  a[2] = (bf16)f0.z;  a[3] = (bf16)f0.w;
    a[4] = (bf16)f1.x;  a[5] = (bf16)f1.y;  a[6] = (bf16)f1.z;  a[7] = (bf16)f1.w;
    a[8] = (bf16)f2.x;  a[9] = (bf16)f2.y;  a[10] = (bf16)f2.z; a[11] = (bf16)f2.w;
    a[12] = (bf16)f3.x; a[13] = (bf16)f3.y; a[14] = (bf16)f3.z; a[15] = (bf16)f3.w;
#pragma unroll
    for (int t = 0; t < 4; ++t) acc[t] = wmma_bf16(a, bfr[t], acc[t]);
  }
#pragma unroll
  for (int t = 0; t < 4; ++t) {
    const float bval = bias[t * 16 + n];
#pragma unroll
    for (int r = 0; r < 8; ++r) {
      const int m = row0 + 8 * h + r;            // C/D layout: M = 8*half + r
      outp[(size_t)m * Dn + t * 16 + n] = (bf16)(acc[t][r] + bval);
    }
  }
}

// ---------------------------------------------------------------------------
// Kernel 3: causal flash attention.  4 waves/WG, 64 query rows per WG,
// 128-column KV blocks (halves softmax/shuffle overhead per WMMA).
// Wave-uniform subtile clipping (tlim/clim) skips fully-masked diagonal work.
// ---------------------------------------------------------------------------
constexpr int SB = 128;                              // KV block columns

__global__ __launch_bounds__(128) void attn_kernel(const bf16* __restrict__ Q,
                                                   const bf16* __restrict__ Km,
                                                   const bf16* __restrict__ Vm,
                                                   float*      __restrict__ out) {
  __shared__ __align__(32) bf16 ldsVt[Dn * SB];      // [h][s]  16 KB
  __shared__ __align__(32) bf16 ldsP[4 * 16 * SB];   // per-wave P tile, 16 KB

  const int tid  = threadIdx.x;
  const int lane = tid & 31;
  const int wave = tid >> 5;
  const int h    = lane >> 4;
  const int n    = lane & 15;
  const int b    = blockIdx.y;
  const int m0   = blockIdx.x * 64;
  const int row0 = m0 + wave * 16;

  // Q A-fragments for the whole pass (K-dim = D = 64 -> two 16x32 chunks).
  v16bf aQ[2];
  const size_t qbase = ((size_t)b * Tn + row0 + n) * Dn;
#pragma unroll
  for (int c = 0; c < 2; ++c) {
    const v8bf r0 = *(const v8bf*)(Q + qbase + c * 32 + 16 * h);
    const v8bf r1 = *(const v8bf*)(Q + qbase + c * 32 + 16 * h + 8);
#pragma unroll
    for (int e = 0; e < 8; ++e) { aQ[c][e] = r0[e]; aQ[c][8 + e] = r1[e]; }
  }

  float mrow[8], lrow[8];
  v8f   acc[4];
#pragma unroll
  for (int r = 0; r < 8; ++r) { mrow[r] = NEGINF; lrow[r] = 0.f; }
#pragma unroll
  for (int t = 0; t < 4; ++t)
#pragma unroll
    for (int r = 0; r < 8; ++r) acc[t][r] = 0.f;

  const int nblk = m0 / SB + 1;                      // causal block range
  for (int jb = 0; jb < nblk; ++jb) {
    const int s0 = jb * SB;
    // Wave-uniform clip: highest S subtile with any unmasked element.
    const bool diag = (jb == nblk - 1);
    const int  tlim = diag ? ((row0 + 15 - s0) / 16 + 1) : 8;   // 1..8
    const int  clim = (tlim + 1) >> 1;                          // PV k-chunks

    __syncthreads();                                 // prior Vt readers done
    // Stage V transposed into LDS: ldsVt[h][s] = V[s0+s][h]
#pragma unroll
    for (int p = 0; p < 8; ++p) {
      const int e8 = p * 128 + tid;                  // 1024 chunks of 8
      const int s  = e8 >> 3;
      const int hb = (e8 & 7) * 8;
      const v8bf v = *(const v8bf*)(Vm + ((size_t)b * Tn + s0 + s) * Dn + hb);
#pragma unroll
      for (int i = 0; i < 8; ++i) ldsVt[(hb + i) * SB + s] = v[i];
    }
    if (jb + 1 < nblk) {                             // hint next K/V block
      __builtin_prefetch(Km + ((size_t)b * Tn + s0 + SB + lane) * Dn, 0, 1);
      __builtin_prefetch(Vm + ((size_t)b * Tn + s0 + SB + lane) * Dn, 0, 1);
    }
    __syncthreads();                                 // Vt visible

    // S = (Q Kt) * scale : up to 8 column subtiles x 2 k-chunks = 16 WMMAs
    float S[8][8];
#pragma unroll
    for (int t = 0; t < 8; ++t) {
      if (t < tlim) {
        v8f sa;
#pragma unroll
        for (int r = 0; r < 8; ++r) sa[r] = 0.f;
#pragma unroll
        for (int c = 0; c < 2; ++c) {
          // B[K=d][N=s] = K[s][d]: contiguous along d in memory
          const bf16* kp =
              Km + ((size_t)b * Tn + s0 + t * 16 + n) * Dn + c * 32 + 16 * h;
          const v16bf bk = *(const v16bf*)kp;
          sa = wmma_bf16(aQ[c], bk, sa);
        }
#pragma unroll
        for (int r = 0; r < 8; ++r) S[t][r] = sa[r] * SCALE;
      } else {
#pragma unroll
        for (int r = 0; r < 8; ++r) S[t][r] = NEGINF;
      }
    }

    // Per-element causal mask, diagonal block only.
    if (diag) {
#pragma unroll
      for (int t = 0; t < 8; ++t)
#pragma unroll
        for (int r = 0; r < 8; ++r) {
          const int mg = row0 + 8 * h + r;
          const int sg = s0 + t * 16 + n;
          if (sg > mg) S[t][r] = NEGINF;
        }
    }

    // Online softmax update (row groups = 16 lanes sharing one half).
    float corr[8];
#pragma unroll
    for (int r = 0; r < 8; ++r) {
      float mx = S[0][r];
#pragma unroll
      for (int t = 1; t < 8; ++t) mx = fmaxf(mx, S[t][r]);
      for (int o = 1; o < 16; o <<= 1) mx = fmaxf(mx, __shfl_xor(mx, o, 16));
      const float mnew = fmaxf(mrow[r], mx);
      corr[r] = exp2f((mrow[r] - mnew) * LOG2E);
      mrow[r] = mnew;
      float ps = 0.f;
#pragma unroll
      for (int t = 0; t < 8; ++t) {
        const float p = exp2f((S[t][r] - mnew) * LOG2E);
        S[t][r] = p;                                 // masked lanes -> 0
        ps += p;
      }
      for (int o = 1; o < 16; o <<= 1) ps += __shfl_xor(ps, o, 16);
      lrow[r] = lrow[r] * corr[r] + ps;
#pragma unroll
      for (int t = 0; t < 4; ++t) acc[t][r] *= corr[r];
    }

    // P (D-layout) -> LDS -> A-layout fragments.  Masked subtiles store 0,
    // so partially-valid PV chunks read well-defined zeros.
    bf16* pb = ldsP + wave * 16 * SB;
#pragma unroll
    for (int t = 0; t < 8; ++t)
#pragma unroll
      for (int r = 0; r < 8; ++r)
        pb[(8 * h + r) * SB + t * 16 + n] = (bf16)S[t][r];
    __syncthreads();                                 // P + ordering visible

    // O += P @ V : up to 4 k-chunks x 4 head subtiles = 16 WMMAs
#pragma unroll
    for (int c = 0; c < 4; ++c) {
      if (c < clim) {
        v16bf aP;
        const bf16* pr = pb + n * SB + c * 32;       // row M = lane%16
        const v8bf r0 = *(const v8bf*)(pr + 16 * h);
        const v8bf r1 = *(const v8bf*)(pr + 16 * h + 8);
#pragma unroll
        for (int e = 0; e < 8; ++e) { aP[e] = r0[e]; aP[8 + e] = r1[e]; }
#pragma unroll
        for (int t = 0; t < 4; ++t) {
          const v16bf bv =
              *(const v16bf*)(ldsVt + (t * 16 + n) * SB + c * 32 + 16 * h);
          acc[t] = wmma_bf16(aP, bv, acc[t]);
        }
      }
    }
  }

  // Epilogue: normalize and store fp32 output.
#pragma unroll
  for (int t = 0; t < 4; ++t)
#pragma unroll
    for (int r = 0; r < 8; ++r) {
      const int m = row0 + 8 * h + r;
      out[((size_t)b * Tn + m) * Dn + t * 16 + n] = acc[t][r] / lrow[r];
    }
}

// ---------------------------------------------------------------------------
extern "C" void kernel_launch(void* const* d_in, const int* in_sizes, int n_in,
                              void* d_out, int out_size, void* d_ws, size_t ws_size,
                              hipStream_t stream) {
  (void)in_sizes; (void)n_in; (void)out_size; (void)ws_size;
  const float* emb = (const float*)d_in[0];
  const float* Wk  = (const float*)d_in[1];
  const float* bk  = (const float*)d_in[2];
  const float* Wq  = (const float*)d_in[3];
  const float* bq  = (const float*)d_in[4];
  const float* Wv  = (const float*)d_in[5];
  const float* bv  = (const float*)d_in[6];
  float* out = (float*)d_out;

  // Workspace layout: [ Wt bf16 3*64*1024 | K,Q,V bf16 3*B*T*64 ]
  bf16* Wt  = (bf16*)d_ws;
  bf16* qkv = (bf16*)((char*)d_ws + (size_t)3 * Dn * En * sizeof(bf16));
  bf16* Kp  = qkv;
  bf16* Qp  = qkv + (size_t)1 * Bn * Tn * Dn;
  bf16* Vp  = qkv + (size_t)2 * Bn * Tn * Dn;

  // 1) weight convert/transpose
  wcvt_kernel<<<(3 * Dn * En + 255) / 256, 256, 0, stream>>>(Wk, Wq, Wv, Wt);
  // 2) projections: grid.x = (B*T/16)/4 wave-tiles per WG, grid.y = matrix
  proj_kernel<<<dim3((Bn * Tn / 16) / 4, 3), 128, 0, stream>>>(emb, bk, bq, bv, Wt, qkv);
  // 3) causal flash attention: grid.x = T/64 row blocks, grid.y = batch
  attn_kernel<<<dim3(Tn / 64, Bn), 128, 0, stream>>>(Qp, Kp, Vp, out);
}
